// MetaPath2Vec_11020886081829
// MI455X (gfx1250) — matrix operand order repaired and checked by reading, not compiled, not run on hardware
//
#include <hip/hip_runtime.h>
#include <hip/hip_bf16.h>

typedef __attribute__((ext_vector_type(16))) _Float16 v16h;
typedef __attribute__((ext_vector_type(8)))  float    v8f;

#define NUM_USERS   100000
#define NUM_ITEMS   50000
#define EMBED_DIM   128
#define WALK_LEN    10
#define CTX_W       5
#define WPN         4
#define NNEG        2
#define BATCH_N     4096
#define START_ITEM  0
#define START_USER  NUM_ITEMS
#define DUMMY_IDX   (NUM_USERS + NUM_ITEMS)
#define NT_POS      (BATCH_N * WPN)          /* 16384 */
#define NT_NEG      (BATCH_N * WPN * NNEG)   /* 32768 */
#define NUM_W       (1 + WALK_LEN + 1 - CTX_W) /* 7 */
#define NPOS_ROWS   (NUM_W * NT_POS)         /* 114688 */
#define NNEG_ROWS   (NUM_W * NT_NEG)         /* 229376 */
#define RW_COLS     (WALK_LEN + 1)           /* 11 */

__device__ __forceinline__ unsigned mix32(unsigned x) {
  x ^= x >> 17; x *= 0xed5ad4bbu;
  x ^= x >> 11; x *= 0xac4c1b51u;
  x ^= x >> 15; x *= 0x31848babu;
  x ^= x >> 14;
  return x;
}
__device__ __forceinline__ unsigned rng_u32(unsigned t, unsigned step, unsigned salt) {
  return mix32(t * 0x9E3779B9u ^ (step + 1u) * 0x85EBCA6Bu ^ salt * 0xC2B2AE35u ^ 0xA511E9B3u);
}

// ---------------- positive walks (CSR alternating user->item->user) ----------
__global__ void walk_pos_kernel(const int* __restrict__ batch,
                                const int* __restrict__ rowptr_ui, const int* __restrict__ col_ui,
                                const int* __restrict__ rowcount_ui, int nnz_ui,
                                const int* __restrict__ rowptr_iu, const int* __restrict__ col_iu,
                                const int* __restrict__ rowcount_iu, int nnz_iu,
                                int* __restrict__ rw) {
  int t = blockIdx.x * blockDim.x + threadIdx.x;
  if (t >= NT_POS) return;
  int cur = batch[t % BATCH_N];
  rw[t * RW_COLS + 0] = min(cur + START_USER, DUMMY_IDX);
  #pragma unroll
  for (int i = 0; i < WALK_LEN; ++i) {
    const bool even = (i % 2 == 0);
    const int* rowptr = even ? rowptr_ui : rowptr_iu;
    const int* colv   = even ? col_ui : col_iu;
    const int* rc     = even ? rowcount_ui : rowcount_iu;
    const int  nsrc   = even ? NUM_USERS : NUM_ITEMS;
    const int  nnz    = even ? nnz_ui : nnz_iu;
    bool mask = (cur >= DUMMY_IDX);
    int s = min(max(cur, 0), nsrc - 1);
    int count = rc[s];
    float r = (float)(rng_u32((unsigned)t, (unsigned)i, 0x1234u) >> 8) * (1.0f / 16777216.0f);
    int idx = (int)(r * (float)count) + rowptr[s];
    idx = min(idx, nnz - 1);
    int nxt = colv[idx];
    cur = (mask || count == 0) ? DUMMY_IDX : nxt;
    int off = even ? START_ITEM : START_USER;
    rw[t * RW_COLS + i + 1] = min(cur + off, DUMMY_IDX);
  }
}

// ---------------- negative walks (uniform random ids) ------------------------
__global__ void walk_neg_kernel(const int* __restrict__ batch, int* __restrict__ rw) {
  int t = blockIdx.x * blockDim.x + threadIdx.x;
  if (t >= NT_NEG) return;
  int cur = batch[t % BATCH_N];
  rw[t * RW_COLS + 0] = min(cur + START_USER, DUMMY_IDX);
  #pragma unroll
  for (int i = 0; i < WALK_LEN; ++i) {
    const bool even = (i % 2 == 0);
    unsigned n_dst = even ? (unsigned)NUM_ITEMS : (unsigned)NUM_USERS;
    unsigned h = rng_u32((unsigned)t, (unsigned)i, 0x9999u);
    cur = (int)(((unsigned long long)h * (unsigned long long)n_dst) >> 32);
    int off = even ? START_ITEM : START_USER;
    rw[t * RW_COLS + i + 1] = min(cur + off, DUMMY_IDX);
  }
}

// ---------------- emit windows into d_out (as floats) ------------------------
__global__ void emit_windows_kernel(const int* __restrict__ rw, int Nt,
                                    float* __restrict__ out) {
  int p = blockIdx.x * blockDim.x + threadIdx.x;
  int nrows = NUM_W * Nt;
  if (p >= nrows) return;
  int w = p / Nt, t = p % Nt;
  #pragma unroll
  for (int c = 0; c < CTX_W; ++c)
    out[(size_t)p * CTX_W + c] = (float)rw[t * RW_COLS + w + c];
}

// ---------------- skip-gram logits via WMMA + loss reduction -----------------
// One wave handles a tile of 16 consecutive window rows. For each context c,
// a 16x16x128 GEMM (4 chained v_wmma_f32_16x16x32_f16) whose diagonal is the
// 16 logits for that context.
__device__ __forceinline__ void cvt8(v16h& v, int base, float4 f0, float4 f1) {
  v[base + 0] = (_Float16)f0.x; v[base + 1] = (_Float16)f0.y;
  v[base + 2] = (_Float16)f0.z; v[base + 3] = (_Float16)f0.w;
  v[base + 4] = (_Float16)f1.x; v[base + 5] = (_Float16)f1.y;
  v[base + 6] = (_Float16)f1.z; v[base + 7] = (_Float16)f1.w;
}

__global__ void __launch_bounds__(256)
skipgram_loss_kernel(const float* __restrict__ emb, const int* __restrict__ rw,
                     int Nt, int negFlag, float* __restrict__ accum) {
  const int lane   = threadIdx.x & 31;
  const int wv     = threadIdx.x >> 5;
  const int tile   = blockIdx.x * (blockDim.x >> 5) + wv;
  const int n0     = tile * 16;
  const int m      = lane & 15;        // window within tile for this lane
  const int p      = n0 + m;
  const int w      = p / Nt;           // uniform across lanes (Nt % 16 == 0)
  const int t      = p % Nt;
  const int hiHalf = lane >> 4;

  // ---- A matrix: start embeddings, 16x32 f16 chunks per ISA layout.
  // lanes 0-15: K = kb+{0..7} (v0-3) and kb+16+{0..7} (v4-7); lanes 16-31: +8.
  const int   sidx = rw[t * RW_COLS + w];
  const float* __restrict__ arow = emb + (size_t)sidx * EMBED_DIM;
  const int asel = hiHalf ? 8 : 0;

  v16h A[4];
  #pragma unroll
  for (int kc = 0; kc < 4; ++kc) {
    const float* p0 = arow + kc * 32 + asel;
    const float* p1 = p0 + 16;
    float4 f0 = *(const float4*)(p0);
    float4 f1 = *(const float4*)(p0 + 4);
    float4 f2 = *(const float4*)(p1);
    float4 f3 = *(const float4*)(p1 + 4);
    v16h a;
    cvt8(a, 0, f0, f1);
    cvt8(a, 8, f2, f3);
    A[kc] = a;
  }

  // ---- B matrix: 32x16 f16 chunk; column N = lane&15; lanes 0-15 hold
  // K = kb+{0..7} and kb+16+{0..7}, lanes 16-31: +8 (matches A).
  const int bsel = hiHalf ? 8 : 0;

  float local = 0.0f;
  const bool active = (lane < 8) || (lane >= 24);
  const int  dsel   = (lane < 8) ? lane : (lane - 24);  // which acc reg holds diag

  #pragma unroll
  for (int c = 0; c < 4; ++c) {
    const int cidx = rw[t * RW_COLS + w + 1 + c];
    const float* __restrict__ brow = emb + (size_t)cidx * EMBED_DIM;
    v8f acc = {};
    #pragma unroll
    for (int kc = 0; kc < 4; ++kc) {
      const float* q = brow + kc * 32 + bsel;
      float4 g0 = *(const float4*)(q);
      float4 g1 = *(const float4*)(q + 4);
      float4 g2 = *(const float4*)(q + 16);
      float4 g3 = *(const float4*)(q + 20);
      v16h b;
      cvt8(b, 0, g0, g1);
      cvt8(b, 8, g2, g3);
      acc = __builtin_amdgcn_wmma_f32_16x16x32_f16(
          /*neg_a=*/false, A[kc], /*neg_b=*/false, b,
          /*c_mod=*/(short)0, acc, /*reuse_a=*/false, /*reuse_b=*/false);
    }
    // D diag: m<8 -> acc[m] @ lane m ; m>=8 -> acc[m-8] @ lane m+16.
    float x = acc[0];
    #pragma unroll
    for (int r2 = 1; r2 < 8; ++r2) x = (dsel == r2) ? acc[r2] : x;
    float sgm = 1.0f / (1.0f + __expf(-x));
    float val = negFlag ? -__logf(1.0f - sgm + 1e-15f)
                        : -__logf(sgm + 1e-15f);
    local += active ? val : 0.0f;
  }

  #pragma unroll
  for (int o2 = 16; o2 > 0; o2 >>= 1) local += __shfl_down(local, o2);
  if (lane == 0) atomicAdd(accum, local);
}

__global__ void init_kernel(float* __restrict__ accum) {
  if (threadIdx.x < 2) accum[threadIdx.x] = 0.0f;
}

__global__ void finalize_kernel(const float* __restrict__ accum, float* __restrict__ out) {
  if (threadIdx.x == 0)
    out[0] = accum[0] / (float)(NPOS_ROWS * 4) + accum[1] / (float)(NNEG_ROWS * 4);
}

extern "C" void kernel_launch(void* const* d_in, const int* in_sizes, int n_in,
                              void* d_out, int out_size, void* d_ws, size_t ws_size,
                              hipStream_t stream) {
  const float* emb         = (const float*)d_in[0];
  const int*   rowptr_ui   = (const int*)d_in[1];
  const int*   col_ui      = (const int*)d_in[2];
  const int*   rowcount_ui = (const int*)d_in[3];
  const int*   rowptr_iu   = (const int*)d_in[4];
  const int*   col_iu      = (const int*)d_in[5];
  const int*   rowcount_iu = (const int*)d_in[6];
  const int*   batch       = (const int*)d_in[7];
  const int nnz_ui = in_sizes[2];
  const int nnz_iu = in_sizes[5];

  float* out    = (float*)d_out;
  float* accum  = (float*)d_ws;                         // 2 partial sums
  int*   rw_pos = (int*)((char*)d_ws + 16);             // 16384 x 11
  int*   rw_neg = rw_pos + NT_POS * RW_COLS;            // 32768 x 11

  init_kernel<<<1, 32, 0, stream>>>(accum);
  walk_pos_kernel<<<NT_POS / 256, 256, 0, stream>>>(
      batch, rowptr_ui, col_ui, rowcount_ui, nnz_ui,
      rowptr_iu, col_iu, rowcount_iu, nnz_iu, rw_pos);
  walk_neg_kernel<<<NT_NEG / 256, 256, 0, stream>>>(batch, rw_neg);

  emit_windows_kernel<<<(NPOS_ROWS + 255) / 256, 256, 0, stream>>>(
      rw_pos, NT_POS, out + 1);
  emit_windows_kernel<<<(NNEG_ROWS + 255) / 256, 256, 0, stream>>>(
      rw_neg, NT_NEG, out + 1 + (size_t)NPOS_ROWS * CTX_W);

  // 16 windows per wave, 8 waves per block; counts divide exactly.
  skipgram_loss_kernel<<<(NPOS_ROWS / 16) / 8, 256, 0, stream>>>(
      emb, rw_pos, NT_POS, 0, accum + 0);
  skipgram_loss_kernel<<<(NNEG_ROWS / 16) / 8, 256, 0, stream>>>(
      emb, rw_neg, NT_NEG, 1, accum + 1);

  finalize_kernel<<<1, 32, 0, stream>>>(accum, out);
}
